// PCUTransformer_53214644797924
// MI455X (gfx1250) — compile-verified, matrix operand order checked by reading, not run on hardware
//
#include <hip/hip_runtime.h>
#include <hip/hip_bf16.h>
#include <math.h>

typedef __attribute__((ext_vector_type(16))) _Float16 v16h;
typedef __attribute__((ext_vector_type(8)))  _Float16 v8h;
typedef __attribute__((ext_vector_type(8)))  float    v8f;

// Model constants
constexpr int B_    = 8;
constexpr int L_    = 1000;
constexpr int F_    = 7;
constexpr int DM_   = 512;
constexpr int H_    = 8;
constexpr int DFF_  = 2048;
constexpr int NL_   = 4;
constexpr int PRED_ = 50;
constexpr int DK_   = 64;
constexpr int P_    = 16;
constexpr int M_    = B_ * L_;     // 8000 tokens (multiple of 32)
constexpr int VLD_  = 1024;        // padded key length for transposed V rows
constexpr float EPS_   = 1e-5f;
constexpr float SCALE_ = 0.125f;   // 1/sqrt(64)

// ---------------------------------------------------------------------------
// WMMA helpers
// ---------------------------------------------------------------------------
__device__ inline v8f wmma_f16(v16h a, v16h b, v8f c) {
  // D = A(16x32 f16) * B(32x16 f16) + C(16x16 f32)
  return __builtin_amdgcn_wmma_f32_16x16x32_f16(false, a, false, b, (short)0, c,
                                                false, false);
}

// Branchless tanh: 1 - 2/(exp(2x)+1). Saturates correctly at +-1 via
// exp overflow/underflow; lowers to v_exp_f32 + v_rcp_f32, no EXEC branches.
__device__ inline float fast_tanh(float x) {
  const float e = __expf(2.0f * x);
  return 1.0f - 2.0f * __builtin_amdgcn_rcpf(e + 1.0f);
}

// Load a 16x32 f16 A/B fragment from a row-major matrix (ld in halves).
// CDNA5 16-bit fragment layout: lane = row (mod 16); halves 0..7 hold
// k = k0+koff .. +7, halves 8..15 hold k = k0+koff+16 .. +23, koff = 8 for
// lanes 16..31. Two contiguous 16B loads per lane.
__device__ inline v16h load_frag(const _Float16* __restrict__ base, int ld,
                                 int row0, int k0, int rowmax) {
  const int lane = threadIdx.x & 31;
  int r = row0 + (lane & 15);
  if (r > rowmax) r = rowmax;
  const int koff = k0 + ((lane >> 4) << 3);
  const _Float16* p = base + (size_t)r * ld + koff;
  v8h lo = *(const v8h*)(p);
  v8h hi = *(const v8h*)(p + 16);
  v16h f;
#pragma unroll
  for (int i = 0; i < 8; ++i) { f[i] = lo[i]; f[i + 8] = hi[i]; }
  return f;
}

// ---------------------------------------------------------------------------
// Elementwise / small kernels
// ---------------------------------------------------------------------------
__global__ void embed_kernel(const float* __restrict__ x,
                             const float* __restrict__ ew,
                             const float* __restrict__ eb,
                             const float* __restrict__ pe,
                             float* __restrict__ h32,
                             _Float16* __restrict__ h16) {
  const int tid = blockIdx.x * blockDim.x + threadIdx.x;
  if (tid >= M_ * DM_) return;
  const int d   = tid & (DM_ - 1);
  const int tok = tid >> 9;  // DM_ == 512
  const int l   = tok % L_;
  float acc = eb[d] + pe[l * DM_ + d];
  const float* xr = x + (size_t)tok * F_;
#pragma unroll
  for (int f = 0; f < F_; ++f) acc += xr[f] * ew[f * DM_ + d];
  h32[tid] = acc;
  h16[tid] = (_Float16)acc;
}

// W[K,N] f32 -> Wt[N,K] f16 (transpose + convert)
__global__ void convt_kernel(const float* __restrict__ W,
                             _Float16* __restrict__ Wt, int K, int N) {
  const int tid = blockIdx.x * blockDim.x + threadIdx.x;
  if (tid >= K * N) return;
  const int k = tid / N, n = tid - k * N;
  Wt[(size_t)n * K + k] = (_Float16)W[tid];
}

// qp/kp projection: src f16 [B,L,DM] -> dst f16 [B,H,L,32] (p>=16 zero-padded)
__global__ void maskproj_kernel(const _Float16* __restrict__ src,
                                const float* __restrict__ w,     // [DK,P]
                                const float* __restrict__ bvec,  // [P]
                                _Float16* __restrict__ dst) {
  const int tid = blockIdx.x * blockDim.x + threadIdx.x;
  if (tid >= B_ * H_ * L_ * 32) return;
  const int p = tid & 31;
  const int l = (tid >> 5) % L_;
  const int h = (tid / (32 * L_)) % H_;
  const int b = tid / (32 * L_ * H_);
  float v = 0.0f;
  if (p < P_) {
    const _Float16* qrow = src + (size_t)(b * L_ + l) * DM_ + h * DK_;
    float acc = bvec[p];
#pragma unroll 8
    for (int k = 0; k < DK_; ++k) acc += (float)qrow[k] * w[k * P_ + p];
    v = acc;
  }
  dst[(size_t)tid] = (_Float16)v;
}

// ---------------------------------------------------------------------------
// WMMA GEMM: C[M,N] = A[M,K] * Bt[N,K]^T + bias.
// One wave per 32x64 tile (2x4 register blocking, 8 WMMAs per K step).
// MODE 0: f32 out   MODE 1: f16 out   MODE 2: f16 out + ReLU
// MODE 3: f16 out scattered to transposed-V layout [B,H,DK,VLD]
// ---------------------------------------------------------------------------
template <int MODE>
__global__ __launch_bounds__(32, 1)
void gemm_wmma_kernel(const _Float16* __restrict__ A,
                      const _Float16* __restrict__ Bt,
                      const float* __restrict__ bias,
                      void* __restrict__ outp,
                      int Mdim, int Ndim, int Kdim) {
  const int n0 = blockIdx.x * 64;
  const int m0 = blockIdx.y * 32;
  const int lane = threadIdx.x & 31;
  const v8f z = {0.f, 0.f, 0.f, 0.f, 0.f, 0.f, 0.f, 0.f};
  v8f c[2][4];
#pragma unroll
  for (int mt = 0; mt < 2; ++mt)
#pragma unroll
    for (int nt = 0; nt < 4; ++nt) c[mt][nt] = z;

  for (int k0 = 0; k0 < Kdim; k0 += 32) {
    v16h a0 = load_frag(A, Kdim, m0, k0, Mdim - 1);
    v16h a1 = load_frag(A, Kdim, m0 + 16, k0, Mdim - 1);
#pragma unroll
    for (int nt = 0; nt < 4; ++nt) {
      v16h b = load_frag(Bt, Kdim, n0 + nt * 16, k0, Ndim - 1);
      c[0][nt] = wmma_f16(a0, b, c[0][nt]);
      c[1][nt] = wmma_f16(a1, b, c[1][nt]);
    }
  }

  auto store_tile = [&](const v8f& cc, int mbase, int nbase) {
    const float bv = bias[nbase + (lane & 15)];
#pragma unroll
    for (int j = 0; j < 8; ++j) {
      const int row = mbase + j + ((lane >> 4) << 3);
      const int col = nbase + (lane & 15);
      float v = cc[j] + bv;
      if (MODE == 2) v = fmaxf(v, 0.0f);
      if (MODE == 0) {
        ((float*)outp)[(size_t)row * Ndim + col] = v;
      } else if (MODE == 1 || MODE == 2) {
        ((_Float16*)outp)[(size_t)row * Ndim + col] = (_Float16)v;
      } else {  // MODE 3: Vt[b, h, dk, l]
        const int bb = row / L_;
        const int l  = row - bb * L_;
        const int hh = col >> 6;
        const int dk = col & 63;
        ((_Float16*)outp)[((size_t)(bb * H_ + hh) * DK_ + dk) * VLD_ + l] =
            (_Float16)v;
      }
    }
  };
#pragma unroll
  for (int mt = 0; mt < 2; ++mt)
#pragma unroll
    for (int nt = 0; nt < 4; ++nt)
      store_tile(c[mt][nt], m0 + mt * 16, n0 + nt * 16);
}

// ---------------------------------------------------------------------------
// Flash attention with learned tanh mask, transposed formulation.
// One wave per 16-query tile per (b,h). We compute S^T = K*Q^T so the
// softmax key-reduction is 8 in-register lanes + one shfl_xor(16), and
// ctx^T = V^T * P^T so the dk axis is contiguous per lane (b128 stores).
// ---------------------------------------------------------------------------
__global__ __launch_bounds__(32, 1)
void attn_kernel(const _Float16* __restrict__ q16,
                 const _Float16* __restrict__ k16,
                 const _Float16* __restrict__ vt,   // [B,H,DK,VLD]
                 const _Float16* __restrict__ qp,   // [B,H,L,32]
                 const _Float16* __restrict__ kp,   // [B,H,L,32]
                 _Float16* __restrict__ ctx16) {    // [B,L,DM]
  __shared__ __align__(16) _Float16 sP[16 * 32];    // [query][key strip of 32]
  const int lane = threadIdx.x & 31;
  const int n    = lane & 15;            // query within tile
  const int hi8  = (lane >> 4) << 3;     // 0 or 8
  const int bh = blockIdx.y;
  const int b = bh / H_, h = bh - b * H_;
  const int q0 = blockIdx.x * 16;

  const _Float16* Qb  = q16 + (size_t)(b * L_) * DM_ + h * DK_;
  const _Float16* Kb  = k16 + (size_t)(b * L_) * DM_ + h * DK_;
  const _Float16* Vb  = vt + (size_t)bh * DK_ * VLD_;
  const _Float16* QPb = qp + (size_t)bh * L_ * 32;
  const _Float16* KPb = kp + (size_t)bh * L_ * 32;

  // B operands (persist across the whole loop): queries
  const v16h bq0 = load_frag(Qb, DM_, q0, 0, L_ - 1);
  const v16h bq1 = load_frag(Qb, DM_, q0, 32, L_ - 1);
  const v16h bqp = load_frag(QPb, 32, q0, 0, L_ - 1);

  const v8f z = {0.f, 0.f, 0.f, 0.f, 0.f, 0.f, 0.f, 0.f};
  float mcol = -1e30f, lcol = 0.0f;      // per-query softmax state
  v8f acc[4];                            // ctx^T: dk tiles x queries
#pragma unroll
  for (int t = 0; t < 4; ++t) acc[t] = z;

  for (int pair = 0; pair < 32; ++pair) {
    const int c0 = pair * 32;
    v8f s[2];
#pragma unroll
    for (int t = 0; t < 2; ++t) {
      const int ct = c0 + t * 16;
      v16h ak0 = load_frag(Kb, DM_, ct, 0, L_ - 1);
      v16h ak1 = load_frag(Kb, DM_, ct, 32, L_ - 1);
      v8f st = wmma_f16(ak0, bq0, z);
      st = wmma_f16(ak1, bq1, st);
      v16h akp = load_frag(KPb, 32, ct, 0, L_ - 1);
      v8f mk = wmma_f16(akp, bqp, z);
#pragma unroll
      for (int j = 0; j < 8; ++j) {
        const int key = ct + hi8 + j;
        const float v = st[j] * SCALE_ + fast_tanh(mk[j]);
        st[j] = (key < L_) ? v : -1e30f;
      }
      s[t] = st;
    }
    // Per-query (per-lane) online softmax: in-register reduce + 1 shuffle.
    float mx = -1e30f;
#pragma unroll
    for (int j = 0; j < 8; ++j) mx = fmaxf(mx, fmaxf(s[0][j], s[1][j]));
    mx = fmaxf(mx, __shfl_xor(mx, 16));
    const float mnew = fmaxf(mcol, mx);
    const float corr = __expf(mcol - mnew);
    float rs = 0.0f;
    v8h p0h, p1h;
#pragma unroll
    for (int j = 0; j < 8; ++j) {
      const float p = __expf(s[0][j] - mnew);
      rs += p;
      p0h[j] = (_Float16)p;
    }
#pragma unroll
    for (int j = 0; j < 8; ++j) {
      const float p = __expf(s[1][j] - mnew);
      rs += p;
      p1h[j] = (_Float16)p;
    }
    rs += __shfl_xor(rs, 16);
    lcol = lcol * corr + rs;
    mcol = mnew;
#pragma unroll
    for (int t = 0; t < 4; ++t)
#pragma unroll
      for (int j = 0; j < 8; ++j) acc[t][j] *= corr;

    // Stage P row-major [query][key] -> two b128 LDS stores per lane.
    *(v8h*)&sP[n * 32 + hi8]      = p0h;
    *(v8h*)&sP[n * 32 + 16 + hi8] = p1h;
    __syncthreads();
    const v16h bp = load_frag((const _Float16*)sP, 32, 0, 0, 15);
#pragma unroll
    for (int t = 0; t < 4; ++t) {
      v16h av = load_frag(Vb, VLD_, t * 16, c0, DK_ - 1);
      acc[t] = wmma_f16(av, bp, acc[t]);
    }
    __syncthreads();
  }

  // ctx^T element (dk = t*16 + hi8 + j, query = n): dk contiguous in j.
  const int q = q0 + n;
  if (q < L_) {
    const float inv = 1.0f / lcol;
    _Float16* orow = ctx16 + (size_t)(b * L_ + q) * DM_ + h * DK_;
#pragma unroll
    for (int t = 0; t < 4; ++t) {
      v8h o;
#pragma unroll
      for (int j = 0; j < 8; ++j) o[j] = (_Float16)(acc[t][j] * inv);
      *(v8h*)&orow[t * 16 + hi8] = o;
    }
  }
}

// ---------------------------------------------------------------------------
// LayerNorm kernels (blockDim == DM_ == 512, one token per block)
// ---------------------------------------------------------------------------
__device__ inline float block_reduce_sum(float v, float* sh) {
#pragma unroll
  for (int d = 16; d >= 1; d >>= 1) v += __shfl_xor(v, d);
  const int lane = threadIdx.x & 31, w = threadIdx.x >> 5;
  if (lane == 0) sh[w] = v;
  __syncthreads();
  float r = 0.0f;
  if (w == 0) {
    r = (lane < (int)(blockDim.x >> 5)) ? sh[lane] : 0.0f;
#pragma unroll
    for (int d = 8; d >= 1; d >>= 1) r += __shfl_xor(r, d);
    if (lane == 0) sh[0] = r;
  }
  __syncthreads();
  r = sh[0];
  __syncthreads();
  return r;
}

__global__ void add_ln_kernel(float* __restrict__ h32,
                              _Float16* __restrict__ h16,
                              const float* __restrict__ add,
                              const float* __restrict__ g,
                              const float* __restrict__ be) {
  __shared__ float sh[32];
  const int row = blockIdx.x;
  const int d = threadIdx.x;
  const size_t idx = (size_t)row * DM_ + d;
  const float x = h32[idx] + add[idx];
  const float mean = block_reduce_sum(x, sh) * (1.0f / DM_);
  const float dx = x - mean;
  const float var = block_reduce_sum(dx * dx, sh) * (1.0f / DM_);
  const float y = dx * rsqrtf(var + EPS_) * g[d] + be[d];
  h32[idx] = y;
  h16[idx] = (_Float16)y;
}

__global__ void double_ln_kernel(float* __restrict__ h32,
                                 _Float16* __restrict__ h16,
                                 const float* __restrict__ y,
                                 const float* __restrict__ g,
                                 const float* __restrict__ be) {
  __shared__ float sh[32];
  const int row = blockIdx.x;
  const int d = threadIdx.x;
  const size_t idx = (size_t)row * DM_ + d;
  const float yy = y[idx];
  const float t = h32[idx] + yy;
  float mean = block_reduce_sum(t, sh) * (1.0f / DM_);
  float dx = t - mean;
  float var = block_reduce_sum(dx * dx, sh) * (1.0f / DM_);
  const float h2 = dx * rsqrtf(var + EPS_) * g[d] + be[d];
  const float u = h2 + yy;
  mean = block_reduce_sum(u, sh) * (1.0f / DM_);
  dx = u - mean;
  var = block_reduce_sum(dx * dx, sh) * (1.0f / DM_);
  const float out = dx * rsqrtf(var + EPS_) * g[d] + be[d];
  h32[idx] = out;
  h16[idx] = (_Float16)out;
}

__global__ void final_kernel(const float* __restrict__ h32,
                             const float* __restrict__ gn,
                             const float* __restrict__ bn,
                             const float* __restrict__ fcw,
                             const float* __restrict__ fcb,
                             float* __restrict__ out) {
  __shared__ float sh[32];
  __shared__ float sn[DM_];
  const int b = blockIdx.x;
  const int d = threadIdx.x;
  const size_t idx = (size_t)(b * L_ + (L_ - 1)) * DM_ + d;
  const float x = h32[idx];
  const float mean = block_reduce_sum(x, sh) * (1.0f / DM_);
  const float dx = x - mean;
  const float var = block_reduce_sum(dx * dx, sh) * (1.0f / DM_);
  sn[d] = dx * rsqrtf(var + EPS_) * gn[d] + bn[d];
  __syncthreads();
  if (d < PRED_) {
    float acc = fcb[d];
    for (int k = 0; k < DM_; ++k) acc += sn[k] * fcw[k * PRED_ + d];
    out[b * PRED_ + d] = acc;
  }
}

// ---------------------------------------------------------------------------
// Host launcher
// ---------------------------------------------------------------------------
extern "C" void kernel_launch(void* const* d_in, const int* in_sizes, int n_in,
                              void* d_out, int out_size, void* d_ws,
                              size_t ws_size, hipStream_t stream) {
  (void)in_sizes; (void)n_in; (void)out_size; (void)ws_size;
  const float* x     = (const float*)d_in[0];
  const float* emb_w = (const float*)d_in[1];
  const float* emb_b = (const float*)d_in[2];
  const float* pe    = (const float*)d_in[3];
  const float* Wq = (const float*)d_in[4];  const float* bq = (const float*)d_in[5];
  const float* Wk = (const float*)d_in[6];  const float* bk = (const float*)d_in[7];
  const float* Wv = (const float*)d_in[8];  const float* bv = (const float*)d_in[9];
  const float* Wo = (const float*)d_in[10]; const float* bo = (const float*)d_in[11];
  const float* W1 = (const float*)d_in[12]; const float* b1 = (const float*)d_in[13];
  const float* W2 = (const float*)d_in[14]; const float* b2 = (const float*)d_in[15];
  const float* g1 = (const float*)d_in[16]; const float* be1 = (const float*)d_in[17];
  const float* g2 = (const float*)d_in[18]; const float* be2 = (const float*)d_in[19];
  const float* lm_qw = (const float*)d_in[20]; const float* lm_qb = (const float*)d_in[21];
  const float* lm_kw = (const float*)d_in[22]; const float* lm_kb = (const float*)d_in[23];
  const float* gn = (const float*)d_in[24]; const float* bn = (const float*)d_in[25];
  const float* fc_w = (const float*)d_in[26]; const float* fc_b = (const float*)d_in[27];
  float* out = (float*)d_out;

  char* ws = (char*)d_ws;
  size_t off = 0;
  auto alloc = [&](size_t bytes) -> void* {
    void* p = ws + off;
    off += (bytes + 255) & ~(size_t)255;
    return p;
  };
  float*    h32   = (float*)alloc((size_t)M_ * DM_ * sizeof(float));
  _Float16* h16   = (_Float16*)alloc((size_t)M_ * DM_ * 2);
  _Float16* q16   = (_Float16*)alloc((size_t)M_ * DM_ * 2);
  _Float16* k16   = (_Float16*)alloc((size_t)M_ * DM_ * 2);
  _Float16* vt16  = (_Float16*)alloc((size_t)B_ * H_ * DK_ * VLD_ * 2);
  _Float16* qp16  = (_Float16*)alloc((size_t)B_ * H_ * L_ * 32 * 2);
  _Float16* kp16  = (_Float16*)alloc((size_t)B_ * H_ * L_ * 32 * 2);
  _Float16* ctx16 = (_Float16*)alloc((size_t)M_ * DM_ * 2);
  float*    tmp32 = (float*)alloc((size_t)M_ * DM_ * sizeof(float));
  _Float16* ffn16 = (_Float16*)alloc((size_t)M_ * DFF_ * 2);
  _Float16* wq_t  = (_Float16*)alloc((size_t)NL_ * DM_ * DM_ * 2);
  _Float16* wk_t  = (_Float16*)alloc((size_t)NL_ * DM_ * DM_ * 2);
  _Float16* wv_t  = (_Float16*)alloc((size_t)NL_ * DM_ * DM_ * 2);
  _Float16* wo_t  = (_Float16*)alloc((size_t)NL_ * DM_ * DM_ * 2);
  _Float16* w1_t  = (_Float16*)alloc((size_t)NL_ * DM_ * DFF_ * 2);
  _Float16* w2_t  = (_Float16*)alloc((size_t)NL_ * DFF_ * DM_ * 2);

  // Embedding + positional encoding, f32 + f16 copies of h.
  embed_kernel<<<(M_ * DM_ + 255) / 256, 256, 0, stream>>>(x, emb_w, emb_b, pe,
                                                           h32, h16);

  // Convert all weights to f16, transposed to [N,K] for B-fragment loads.
  const int gDD = (DM_ * DM_ + 255) / 256;
  const int gDF = (DM_ * DFF_ + 255) / 256;
  for (int i = 0; i < NL_; ++i) {
    const size_t oDD = (size_t)i * DM_ * DM_;
    const size_t oDF = (size_t)i * DM_ * DFF_;
    convt_kernel<<<gDD, 256, 0, stream>>>(Wq + oDD, wq_t + oDD, DM_, DM_);
    convt_kernel<<<gDD, 256, 0, stream>>>(Wk + oDD, wk_t + oDD, DM_, DM_);
    convt_kernel<<<gDD, 256, 0, stream>>>(Wv + oDD, wv_t + oDD, DM_, DM_);
    convt_kernel<<<gDD, 256, 0, stream>>>(Wo + oDD, wo_t + oDD, DM_, DM_);
    convt_kernel<<<gDF, 256, 0, stream>>>(W1 + oDF, w1_t + oDF, DM_, DFF_);
    convt_kernel<<<gDF, 256, 0, stream>>>(W2 + oDF, w2_t + oDF, DFF_, DM_);
  }

  const dim3 gemmDD(DM_ / 64, M_ / 32);   // 32x64 tiles
  const dim3 gemmDF(DFF_ / 64, M_ / 32);
  const int QT = (L_ + 15) / 16;          // 63 query tiles
  const dim3 attnGrid(QT, B_ * H_);
  const int mpTot = B_ * H_ * L_ * 32;

  for (int i = 0; i < NL_; ++i) {
    const size_t oDD = (size_t)i * DM_ * DM_;
    const size_t oDF = (size_t)i * DM_ * DFF_;

    // Q/K/V projections (WMMA f16)
    gemm_wmma_kernel<1><<<gemmDD, 32, 0, stream>>>(h16, wq_t + oDD, bq + i * DM_,
                                                   q16, M_, DM_, DM_);
    gemm_wmma_kernel<1><<<gemmDD, 32, 0, stream>>>(h16, wk_t + oDD, bk + i * DM_,
                                                   k16, M_, DM_, DM_);
    gemm_wmma_kernel<3><<<gemmDD, 32, 0, stream>>>(h16, wv_t + oDD, bv + i * DM_,
                                                   vt16, M_, DM_, DM_);
    // Learned-mask projections (K-padded to 32 with zeros)
    maskproj_kernel<<<(mpTot + 255) / 256, 256, 0, stream>>>(q16, lm_qw, lm_qb,
                                                             qp16);
    maskproj_kernel<<<(mpTot + 255) / 256, 256, 0, stream>>>(k16, lm_kw, lm_kb,
                                                             kp16);
    // Flash attention with tanh learned mask
    attn_kernel<<<attnGrid, 32, 0, stream>>>(q16, k16, vt16, qp16, kp16, ctx16);
    // Output projection, residual + LN1
    gemm_wmma_kernel<0><<<gemmDD, 32, 0, stream>>>(ctx16, wo_t + oDD, bo + i * DM_,
                                                   tmp32, M_, DM_, DM_);
    add_ln_kernel<<<M_, DM_, 0, stream>>>(h32, h16, tmp32, g1 + i * DM_,
                                          be1 + i * DM_);
    // FFN: relu(h@W1+b1)@W2+b2, then the double LayerNorm
    gemm_wmma_kernel<2><<<gemmDF, 32, 0, stream>>>(h16, w1_t + oDF, b1 + i * DFF_,
                                                   ffn16, M_, DFF_, DM_);
    gemm_wmma_kernel<0><<<gemmDD, 32, 0, stream>>>(ffn16, w2_t + oDF, b2 + i * DM_,
                                                   tmp32, M_, DM_, DFF_);
    double_ln_kernel<<<M_, DM_, 0, stream>>>(h32, h16, tmp32, g2 + i * DM_,
                                             be2 + i * DM_);
  }

  final_kernel<<<B_, DM_, 0, stream>>>(h32, gn, bn, fc_w, fc_b, out);
}
